// PerDeviceMamba_53876069761153
// MI455X (gfx1250) — matrix operand order repaired and checked by reading, not compiled, not run on hardware
//
#include <hip/hip_runtime.h>

// ---------------------------------------------------------------------------
// Mamba2 forward (B=8, L=2048, 4 layers) for gfx1250 (MI455X, wave32, WMMA).
// GEMMs: v_wmma_f32_16x16x32_bf16, async global->LDS staging, double-buffered.
// ---------------------------------------------------------------------------

typedef __attribute__((ext_vector_type(8)))  float  v8f;
typedef __attribute__((ext_vector_type(16))) __bf16 v16bf;

union FragBF16 { v16bf v; uint4 q[2]; };

// ---- problem constants ----------------------------------------------------
#define B_    8
#define L_    2048
#define DM    512          // d_model
#define DI    1024         // d_inner
#define DSTATE 64
#define NH    16           // heads
#define HD    64           // headdim
#define CD    1152         // conv dim = DI + 2*DSTATE
#define DP    2192         // d_proj = 2*DI + 2*DSTATE + NH
#define DPAD  2304         // DP padded to multiple of 128 (GEMM1 N-tile)
#define NL    4
#define M_    (B_ * L_)    // 16384 rows

// ---- workspace layout (bytes) ---------------------------------------------
#define OFF_X   ((size_t)0)            // x f32  [M, DM]             33.6 MB
#define OFF_XN  ((size_t)33554432)     // xn bf16 [M, DM]            16.8 MB
#define OFF_ZX  ((size_t)50331648)     // zxbcdt f32 [M, DPAD]      151.0 MB
#define OFF_XC  ((size_t)201326592)    // conv(xBC) f32 [M, CD]      75.5 MB
#define OFF_YG  OFF_XC                 // gated-y bf16 [M, DI] (aliases XC)
#define OFF_W1  ((size_t)276824064)    // WT1 bf16 [NL, DPAD, DM]     9.4 MB
#define OFF_W2  ((size_t)286261248)    // WT2 bf16 [NL, DM, DI]       4.2 MB
#define WS_NEED ((size_t)290455552)

// CDNA5 async global->LDS copy (ASYNCcnt) --------------------------------
#define ASYNC_LD16(ldsoff, gaddr)                                           \
    asm volatile("global_load_async_to_lds_b128 %0, %1, off"                \
                 :: "v"(ldsoff), "v"(gaddr) : "memory")
#define WAIT_ASYNC() asm volatile("s_wait_asynccnt 0x0" ::: "memory")

__device__ __forceinline__ float sigmoidf_(float x) { return 1.0f / (1.0f + __expf(-x)); }
__device__ __forceinline__ float siluf_(float x)    { return x * sigmoidf_(x); }
__device__ __forceinline__ float softplusf_(float x){ return (x > 20.f) ? x : log1pf(__expf(x)); }

// ---------------------------------------------------------------------------
// in_proj_w (NL, DM, DP) -> WT1 (NL, DPAD, DM), transposed, bf16, zero-padded
// ---------------------------------------------------------------------------
__global__ void cvt_w1_kernel(const float* __restrict__ W, __bf16* __restrict__ WT) {
    const size_t total = (size_t)NL * DPAD * DM;
    for (size_t i = blockIdx.x * blockDim.x + threadIdx.x; i < total;
         i += (size_t)gridDim.x * blockDim.x) {
        int k = (int)(i & (DM - 1));
        int n = (int)((i >> 9) % DPAD);
        int l = (int)(i / ((size_t)DPAD * DM));
        float v = (n < DP) ? W[(size_t)l * DM * DP + (size_t)k * DP + n] : 0.0f;
        WT[i] = (__bf16)v;
    }
}

// out_proj_w (NL, DI, DM) -> WT2 (NL, DM, DI)
__global__ void cvt_w2_kernel(const float* __restrict__ W, __bf16* __restrict__ WT) {
    const size_t total = (size_t)NL * DM * DI;
    for (size_t i = blockIdx.x * blockDim.x + threadIdx.x; i < total;
         i += (size_t)gridDim.x * blockDim.x) {
        int k = (int)(i & (DI - 1));
        int n = (int)((i >> 10) & (DM - 1));
        int l = (int)(i >> 19);
        WT[i] = (__bf16)W[(size_t)l * DI * DM + (size_t)k * DM + n];
    }
}

// ---------------------------------------------------------------------------
// tok = embed_w[x_chunk]; tok[:,0] += h_prev
// ---------------------------------------------------------------------------
__global__ void embed_kernel(const int* __restrict__ tok, const float* __restrict__ EW,
                             const float* __restrict__ hprev, float* __restrict__ X) {
    const size_t total = (size_t)M_ * DM;
    for (size_t i = blockIdx.x * blockDim.x + threadIdx.x; i < total;
         i += (size_t)gridDim.x * blockDim.x) {
        int d = (int)(i & (DM - 1));
        int l = (int)((i >> 9) & (L_ - 1));
        int b = (int)(i / ((size_t)L_ * DM));
        int id = tok[b * L_ + l];
        float v = EW[(size_t)id * DM + d];
        if (l == 0) v += hprev[b * DM + d];
        X[i] = v;
    }
}

// ---------------------------------------------------------------------------
// xn = rms(x, norm_w) + feat @ feat_w   ->  bf16  (one block per (b,l) row)
// ---------------------------------------------------------------------------
__global__ __launch_bounds__(128)
void rmsfeat_kernel(const float* __restrict__ X, const float* __restrict__ nw,
                    const float* __restrict__ fw, const float* __restrict__ feat,
                    __bf16* __restrict__ XN) {
    const int row = blockIdx.x;             // b*L + l
    const int b   = row >> 11;
    const int tid = threadIdx.x;
    const float* xr = X + (size_t)row * DM;
    float v[4]; float ss = 0.f;
#pragma unroll
    for (int i = 0; i < 4; ++i) { v[i] = xr[tid + i * 128]; ss += v[i] * v[i]; }
    __shared__ float red[128];
    red[tid] = ss; __syncthreads();
    for (int s = 64; s > 0; s >>= 1) { if (tid < s) red[tid] += red[tid + s]; __syncthreads(); }
    const float scale = rsqrtf(red[0] * (1.0f / DM) + 1e-5f);
    const float f0 = feat[b*4+0], f1 = feat[b*4+1], f2 = feat[b*4+2], f3 = feat[b*4+3];
#pragma unroll
    for (int i = 0; i < 4; ++i) {
        int c = tid + i * 128;
        float fp = f0*fw[c] + f1*fw[DM+c] + f2*fw[2*DM+c] + f3*fw[3*DM+c];
        XN[(size_t)row * DM + c] = (__bf16)(v[i] * scale * nw[c] + fp);
    }
}

// ---------------------------------------------------------------------------
// GEMM: C[M,N] (+resid) = A[M,K](bf16,row-major) x BT[N,K](bf16,N-major)
// Block 128 thr = 4 waves; block tile 64x128; wave tile 32x64 (2x4 WMMA).
// Double-buffered LDS, async global->LDS copies overlap with WMMA.
// Per-slab addresses advance by +64 B (precomputed pointers, no re-derivation).
// ---------------------------------------------------------------------------
__global__ __launch_bounds__(128)
void gemm_bf16_kernel(const __bf16* __restrict__ A, const __bf16* __restrict__ BT,
                      float* C, int K, int ldc, const float* resid) {
    __shared__ __align__(16) __bf16 sA[2][64 * 40];    // 64 rows(M) x 32(K), stride 40
    __shared__ __align__(16) __bf16 sB[2][128 * 40];   // 128 rows(N) x 32(K), stride 40

    const int tid  = threadIdx.x;
    const int lane = tid & 31;
    const int wave = tid >> 5;
    const int m0 = blockIdx.y * 64;
    const int n0 = blockIdx.x * 128;
    const int wm = (wave >> 1) * 32;
    const int wn = (wave & 1) * 64;
    const int r16 = lane & 15;
    const int kh  = lane >> 4;

    v8f acc[2][4];
#pragma unroll
    for (int i = 0; i < 2; ++i)
#pragma unroll
        for (int j = 0; j < 4; ++j) acc[i][j] = (v8f)0.0f;

    // Precompute per-thread chunk addresses (slab 0); advance by 64 B per slab.
    unsigned long long gA[2], gB[4];
    unsigned int ldsA[2], ldsB[4];
#pragma unroll
    for (int t = 0; t < 2; ++t) {                      // A: 256 chunks / 128 thr
        int c = tid + t * 128;
        int row = c >> 2, cc = c & 3;
        gA[t] = (unsigned long long)(uintptr_t)(A + (size_t)(m0 + row) * K + cc * 8);
        ldsA[t] = (unsigned int)(uintptr_t)(&sA[0][0]) + row * 80 + cc * 16;
    }
#pragma unroll
    for (int t = 0; t < 4; ++t) {                      // B: 512 chunks / 128 thr
        int c = tid + t * 128;
        int row = c >> 2, cc = c & 3;
        gB[t] = (unsigned long long)(uintptr_t)(BT + (size_t)(n0 + row) * K + cc * 8);
        ldsB[t] = (unsigned int)(uintptr_t)(&sB[0][0]) + row * 80 + cc * 16;
    }

    auto issue = [&](int buf) {                        // one 32-wide K slab
        const unsigned int offA = buf ? (unsigned int)sizeof(sA[0]) : 0u;   // 5120
        const unsigned int offB = buf ? (unsigned int)sizeof(sB[0]) : 0u;   // 10240
#pragma unroll
        for (int t = 0; t < 2; ++t) { ASYNC_LD16(ldsA[t] + offA, gA[t]); gA[t] += 64ull; }
#pragma unroll
        for (int t = 0; t < 4; ++t) { ASYNC_LD16(ldsB[t] + offB, gB[t]); gB[t] += 64ull; }
    };

    int cur = 0;
    issue(0);
    WAIT_ASYNC();
    __syncthreads();

    for (int kk = 0; kk < K; kk += 32) {
        if (kk + 32 < K) issue(cur ^ 1);               // overlap copy with WMMA

        FragBF16 fa[2], fb[4];
#pragma unroll
        for (int i = 0; i < 2; ++i) {                  // A: k-halves {kh*8, 16+kh*8}
            const __bf16* pa = &sA[cur][(wm + i * 16 + r16) * 40 + kh * 8];
            fa[i].q[0] = *(const uint4*)pa;
            fa[i].q[1] = *(const uint4*)(pa + 16);
        }
#pragma unroll
        for (int j = 0; j < 4; ++j) {                  // B: k-halves {kh*8, 16+kh*8}
            const __bf16* pb = &sB[cur][(wn + j * 16 + r16) * 40 + kh * 8];
            fb[j].q[0] = *(const uint4*)pb;
            fb[j].q[1] = *(const uint4*)(pb + 16);
        }
#pragma unroll
        for (int i = 0; i < 2; ++i)
#pragma unroll
            for (int j = 0; j < 4; ++j)
                acc[i][j] = __builtin_amdgcn_wmma_f32_16x16x32_bf16(
                    false, fa[i].v, false, fb[j].v, (short)0, acc[i][j], false, false);

        WAIT_ASYNC();          // next slab landed in LDS (own wave's copies)
        __syncthreads();       // ...and everyone's copies / reads complete
        cur ^= 1;
    }

#pragma unroll
    for (int i = 0; i < 2; ++i)
#pragma unroll
        for (int j = 0; j < 4; ++j)
#pragma unroll
            for (int r = 0; r < 8; ++r) {
                int row = m0 + wm + i * 16 + kh * 8 + r;
                int col = n0 + wn + j * 16 + r16;
                size_t idx = (size_t)row * ldc + col;
                float v = acc[i][j][r];
                if (resid) v += resid[idx];
                C[idx] = v;
            }
}

// ---------------------------------------------------------------------------
// Depthwise causal conv (4 taps) + SiLU over xBC cols [DI, DI+CD) of ZX.
// ---------------------------------------------------------------------------
__global__ void conv_silu_kernel(const float* __restrict__ ZX, const float* __restrict__ cw,
                                 const float* __restrict__ cb, float* __restrict__ XC) {
    const size_t total = (size_t)M_ * CD;
    for (size_t i = blockIdx.x * blockDim.x + threadIdx.x; i < total;
         i += (size_t)gridDim.x * blockDim.x) {
        int c = (int)(i % CD);
        int l = (int)((i / CD) & (L_ - 1));
        int b = (int)(i / ((size_t)CD * L_));
        size_t base = (size_t)b * L_ * DPAD;
        float acc = cb[c];
#pragma unroll
        for (int t = 0; t < 4; ++t) {
            int ls = l - 3 + t;
            if (ls >= 0) acc += ZX[base + (size_t)ls * DPAD + DI + c] * cw[t * CD + c];
        }
        XC[i] = siluf_(acc);
    }
}

// ---------------------------------------------------------------------------
// Selective scan. 128 blocks (b,head) x 64 lanes (p). Lane owns h[p, 0:64] in
// VGPRs; B/C broadcast via double-buffered LDS (one barrier/step); next step's
// global loads issued before the FMA loop. y -> ZX cols [DI, 2*DI) (dead xBC).
// ---------------------------------------------------------------------------
__global__ __launch_bounds__(64)
void scan_kernel(float* __restrict__ ZX, const float* __restrict__ XC,
                 const float* __restrict__ dtb, const float* __restrict__ Alog,
                 const float* __restrict__ Dw) {
    const int bh = blockIdx.x;
    const int b  = bh >> 4;
    const int h  = bh & 15;
    const int p  = threadIdx.x;
    __shared__ __align__(16) float sBC[2][128];   // [0:64)=B, [64:128)=C

    float hreg[DSTATE];
#pragma unroll
    for (int n = 0; n < DSTATE; ++n) hreg[n] = 0.f;

    const float Ac   = -__expf(Alog[h]);
    const float bias = dtb[h];
    const float dco  = Dw[h];
    const size_t baseZ = (size_t)b * L_ * DPAD;
    const size_t baseX = (size_t)b * L_ * CD;

    // prologue: stage t = 0
    float dtr_c = ZX[baseZ + 2 * DI + 2 * DSTATE + h];
    float xp_c  = XC[baseX + h * HD + p];
    sBC[0][p]      = XC[baseX + DI + p];
    sBC[0][64 + p] = XC[baseX + DI + DSTATE + p];
    __syncthreads();

    int cb = 0;
    for (int t = 0; t < L_; ++t) {
        float dtr_n = 0.f, xp_n = 0.f;
        if (t + 1 < L_) {                               // pipeline next step
            const size_t ozn = baseZ + (size_t)(t + 1) * DPAD;
            const size_t oxn = baseX + (size_t)(t + 1) * CD;
            dtr_n = ZX[ozn + 2 * DI + 2 * DSTATE + h];
            xp_n  = XC[oxn + h * HD + p];
            sBC[cb ^ 1][p]      = XC[oxn + DI + p];
            sBC[cb ^ 1][64 + p] = XC[oxn + DI + DSTATE + p];
        }

        const float dt  = softplusf_(dtr_c + bias);
        const float dA  = __expf(dt * Ac);
        const float dtx = dt * xp_c;
        const float4* pB = (const float4*)&sBC[cb][0];
        const float4* pC = (const float4*)&sBC[cb][64];
        float y0 = 0.f, y1 = 0.f, y2 = 0.f, y3 = 0.f;
#pragma unroll
        for (int n4 = 0; n4 < 16; ++n4) {
            float4 bb = pB[n4], cc = pC[n4];
            const int n = n4 * 4;
            hreg[n+0] = fmaf(hreg[n+0], dA, dtx * bb.x); y0 = fmaf(hreg[n+0], cc.x, y0);
            hreg[n+1] = fmaf(hreg[n+1], dA, dtx * bb.y); y1 = fmaf(hreg[n+1], cc.y, y1);
            hreg[n+2] = fmaf(hreg[n+2], dA, dtx * bb.z); y2 = fmaf(hreg[n+2], cc.z, y2);
            hreg[n+3] = fmaf(hreg[n+3], dA, dtx * bb.w); y3 = fmaf(hreg[n+3], cc.w, y3);
        }
        ZX[baseZ + (size_t)t * DPAD + DI + h * HD + p] =
            fmaf(dco, xp_c, (y0 + y1) + (y2 + y3));

        __syncthreads();                                // one barrier per step
        dtr_c = dtr_n; xp_c = xp_n; cb ^= 1;
    }
}

// ---------------------------------------------------------------------------
// y = rms(y * silu(z), gnw) -> bf16 for the output GEMM. One block per row.
// ---------------------------------------------------------------------------
__global__ __launch_bounds__(256)
void gate_rms_kernel(const float* __restrict__ ZX, const float* __restrict__ gnw,
                     __bf16* __restrict__ YG) {
    const int row = blockIdx.x;
    const int tid = threadIdx.x;
    const float* zr = ZX + (size_t)row * DPAD;
    float yg[4]; float ss = 0.f;
#pragma unroll
    for (int i = 0; i < 4; ++i) {
        int c = tid + i * 256;
        float z = zr[c];
        float g = zr[DI + c] * siluf_(z);
        yg[i] = g; ss += g * g;
    }
    __shared__ float red[256];
    red[tid] = ss; __syncthreads();
    for (int s = 128; s > 0; s >>= 1) { if (tid < s) red[tid] += red[tid + s]; __syncthreads(); }
    const float scale = rsqrtf(red[0] * (1.0f / DI) + 1e-5f);
#pragma unroll
    for (int i = 0; i < 4; ++i) {
        int c = tid + i * 256;
        YG[(size_t)row * DI + c] = (__bf16)(yg[i] * scale * gnw[c]);
    }
}

// ---------------------------------------------------------------------------
// logits = x @ head_w + head_b   (N=15, tiny GEMV per row)
// ---------------------------------------------------------------------------
__global__ __launch_bounds__(64)
void head_kernel(const float* __restrict__ X, const float* __restrict__ hw,
                 const float* __restrict__ hb, float* __restrict__ out) {
    const int row = blockIdx.x;
    const int tid = threadIdx.x;
    __shared__ float sx[DM];
    __shared__ float red[64];
#pragma unroll
    for (int i = 0; i < 8; ++i) sx[tid + i * 64] = X[(size_t)row * DM + tid + i * 64];
    __syncthreads();
    const int c = tid >> 2, part = tid & 3;
    float sum = 0.f;
    if (c < 15) {
        for (int d = part * 128; d < part * 128 + 128; ++d) sum += sx[d] * hw[d * 15 + c];
    }
    red[tid] = sum; __syncthreads();
    if (tid < 15)
        out[(size_t)row * 15 + tid] =
            red[tid * 4] + red[tid * 4 + 1] + red[tid * 4 + 2] + red[tid * 4 + 3] + hb[tid];
}

__global__ void hnext_kernel(const float* __restrict__ X, float* __restrict__ out) {
    int i = blockIdx.x * blockDim.x + threadIdx.x;
    if (i < B_ * DM) {
        int b = i >> 9, d = i & (DM - 1);
        out[i] = X[((size_t)b * L_ + (L_ - 1)) * DM + d];
    }
}

// ---------------------------------------------------------------------------
extern "C" void kernel_launch(void* const* d_in, const int* in_sizes, int n_in,
                              void* d_out, int out_size, void* d_ws, size_t ws_size,
                              hipStream_t stream) {
    (void)in_sizes; (void)n_in; (void)out_size;
    if (ws_size < WS_NEED) return;   // workspace too small; bail deterministically

    const int*   x_chunk    = (const int*)  d_in[0];
    const float* feat       = (const float*)d_in[1];
    const float* h_prev     = (const float*)d_in[2];
    const float* embed_w    = (const float*)d_in[3];
    const float* norm_w     = (const float*)d_in[4];
    const float* feat_w     = (const float*)d_in[5];
    const float* in_proj_w  = (const float*)d_in[6];
    const float* conv_w     = (const float*)d_in[7];
    const float* conv_b     = (const float*)d_in[8];
    const float* dt_bias    = (const float*)d_in[9];
    const float* A_log      = (const float*)d_in[10];
    const float* Dw         = (const float*)d_in[11];
    const float* gnw        = (const float*)d_in[12];
    const float* out_proj_w = (const float*)d_in[13];
    const float* head_w     = (const float*)d_in[14];
    const float* head_b     = (const float*)d_in[15];
    float* out = (float*)d_out;

    char* ws = (char*)d_ws;
    float*  X   = (float*) (ws + OFF_X);
    __bf16* XN  = (__bf16*)(ws + OFF_XN);
    float*  ZX  = (float*) (ws + OFF_ZX);
    float*  XC  = (float*) (ws + OFF_XC);
    __bf16* YG  = (__bf16*)(ws + OFF_YG);   // aliases XC (XC dead before gating)
    __bf16* WT1 = (__bf16*)(ws + OFF_W1);
    __bf16* WT2 = (__bf16*)(ws + OFF_W2);

    cvt_w1_kernel<<<4096, 256, 0, stream>>>(in_proj_w, WT1);
    cvt_w2_kernel<<<2048, 256, 0, stream>>>(out_proj_w, WT2);
    embed_kernel <<<8192, 256, 0, stream>>>(x_chunk, embed_w, h_prev, X);

    for (int l = 0; l < NL; ++l) {
        rmsfeat_kernel<<<M_, 128, 0, stream>>>(X, norm_w + l * DM,
                                               feat_w + (size_t)l * 4 * DM, feat, XN);
        gemm_bf16_kernel<<<dim3(DPAD / 128, M_ / 64), 128, 0, stream>>>(
            XN, WT1 + (size_t)l * DPAD * DM, ZX, DM, DPAD, nullptr);
        conv_silu_kernel<<<16384, 256, 0, stream>>>(ZX, conv_w + (size_t)l * 4 * CD,
                                                    conv_b + (size_t)l * CD, XC);
        scan_kernel<<<B_ * NH, 64, 0, stream>>>(ZX, XC, dt_bias + l * NH,
                                                A_log + l * NH, Dw + l * NH);
        gate_rms_kernel<<<M_, 256, 0, stream>>>(ZX, gnw + (size_t)l * DI, YG);
        gemm_bf16_kernel<<<dim3(DM / 128, M_ / 64), 128, 0, stream>>>(
            YG, WT2 + (size_t)l * DM * DI, X, DI, DM, /*resid=*/X);
    }

    head_kernel <<<M_, 64, 0, stream>>>(X, head_w, head_b, out);
    hnext_kernel<<<(B_ * DM + 255) / 256, 256, 0, stream>>>(X, out + (size_t)M_ * 15);
}
